// UserContextAttentionPooler_16114717295304
// MI455X (gfx1250) — matrix operand order repaired and checked by reading, not compiled, not run on hardware
//
#include <hip/hip_runtime.h>

// ---------------------------------------------------------------------------
// UserContextAttentionPooler for MI455X (gfx1250, wave32, WMMA).
//
// Shapes: U=128, I=256, J=512, C=E=128.
// Memory floor ~140MB / 23.3TB/s ~= 6us. bf16 WMMA 16x16x32 keeps the two
// GEMMs (attn@interacted, pooled@W2) far under that floor (~330K wmma ops);
// softmax statistics stay in f32 (flash-attention-style precision split).
// ---------------------------------------------------------------------------

typedef __bf16 bf16;
typedef __attribute__((ext_vector_type(16))) __bf16 v16bf;
typedef __attribute__((ext_vector_type(8)))  float  v8f;

#define U_ 128
#define I_ 256
#define J_ 512
#define C_ 128
#define E_ 128

// LDS strides (bf16 elements), padded so row-stride in dwords ≡ 4 (mod 64):
// consecutive lanes (rows) land on disjoint 4-dword bank groups for b128 reads.
#define JLD  520   // interacted^T tile row stride
#define W2LD 136   // W2^T row stride
#define ALD  520   // per-wave attn tile row stride
#define PLD  136   // per-wave pooled tile row stride (aliases attn region)

#define OFF_BT    0
#define SZ_BT    (C_ * JLD * 2)            // 133120 B : bf16 interacted^T [c][j]
#define OFF_W2T  (OFF_BT + SZ_BT)
#define SZ_W2T   (C_ * W2LD * 2)           //  34816 B : bf16 W2^T [n][k]
#define OFF_ATTN (OFF_W2T + SZ_W2T)
#define SZ_ATTN  (8 * 16 * ALD * 2)        // 133120 B : 8 waves x (16 x ALD) bf16
#define OFF_SK   (OFF_ATTN + SZ_ATTN)      //   2048 B : f32 s_k[j]
#define OFF_MADD (OFF_SK + J_ * 4)         //   2048 B : f32 mask add {0,-1e7}
#define OFF_UPB  (OFF_MADD + J_ * 4)       //    512 B : f32 user@W1 + b_mlp
#define OFF_ST   (OFF_UPB + C_ * 4)        //    512 B : f32 s_t for 128 rows
#define SMEM_BYTES (OFF_ST + 128 * 4)      // 306176 B total (< 320KB WGP LDS)

union Frag { uint4 q[2]; v16bf v; };

// No ftanh libcall for gfx1250: build tanh from the native v_exp_f32.
// tanh(x) = sign(x) * (1 - e^{-2|x|}) / (1 + e^{-2|x|})  (stable for all x).
static __device__ __forceinline__ float fast_tanhf(float x) {
  const float ax = __builtin_fabsf(x);
  const float t  = __expf(-2.0f * ax);
  const float r  = (1.0f - t) / (1.0f + t);
  return __builtin_copysignf(r, x);
}

__global__ void __launch_bounds__(256)
ucap_kernel(const float* __restrict__ target,          // (U,I,C)
            const float* __restrict__ inter,           // (U,J,C)
            const float* __restrict__ user,            // (U,E)
            const unsigned char* __restrict__ mask,    // (U,J) bool
            const float* __restrict__ w_dense,         // (2C,)
            const float* __restrict__ b_dense,         // (1,)
            const float* __restrict__ W_mlp,           // (E+C, C)
            const float* __restrict__ b_mlp,           // (C,)
            float* __restrict__ out_ctx,               // (U,I,C)
            float* __restrict__ out_attn) {            // (U,I,J)
  extern __shared__ char smem[];
  bf16*  Bt   = (bf16*)(smem + OFF_BT);
  bf16*  W2t  = (bf16*)(smem + OFF_W2T);
  bf16*  Attn = (bf16*)(smem + OFF_ATTN);
  float* skv  = (float*)(smem + OFF_SK);
  float* madd = (float*)(smem + OFF_MADD);
  float* upb  = (float*)(smem + OFF_UPB);
  float* stv  = (float*)(smem + OFF_ST);

  const int tid   = threadIdx.x;
  const int u     = blockIdx.x >> 1;
  const int ibase = (blockIdx.x & 1) * 128;   // this block owns 128 i-rows

  // ---- Phase A: s_k[j] = interacted[u,j,:] . w_dense[C:], mask bias --------
  for (int j = tid; j < J_; j += 256) {
    const float* p = inter + ((size_t)u * J_ + j) * C_;
    float s = 0.f;
#pragma unroll 8
    for (int c = 0; c < C_; ++c) s += p[c] * w_dense[C_ + c];
    skv[j]  = s;
    madd[j] = mask[u * J_ + j] ? 0.f : -10000000.0f;
  }
  // ---- Phase B: stage interacted[u] transposed in LDS as bf16 (K-contig) ---
  for (int e = tid; e < J_ * C_; e += 256) {
    const int j = e >> 7, c = e & (C_ - 1);
    Bt[c * JLD + j] = (bf16)inter[((size_t)u * J_ + j) * C_ + c];
  }
  // ---- Phase C: stage W2^T = W_mlp[E:,:]^T as bf16 -------------------------
  for (int e = tid; e < C_ * C_; e += 256) {
    const int k = e >> 7, n = e & 127;
    W2t[n * W2LD + k] = (bf16)W_mlp[(E_ + k) * C_ + n];
  }
  // ---- Phase D: upb[n] = user[u]@W_mlp[:E,n] + b_mlp[n] --------------------
  if (tid < C_) {
    float s = b_mlp[tid];
    for (int e2 = 0; e2 < E_; ++e2) s += user[u * E_ + e2] * W_mlp[e2 * C_ + tid];
    upb[tid] = s;
  }
  // ---- Phase E: s_t for this block's 128 target rows -----------------------
  if (tid < 128) {
    const float* p = target + ((size_t)u * I_ + ibase + tid) * C_;
    float s = 0.f;
#pragma unroll 8
    for (int c = 0; c < C_; ++c) s += p[c] * w_dense[c];
    stv[tid] = s;
  }
  __syncthreads();

  // ---- Per-wave: one 16-row i-tile ----------------------------------------
  const int wave = tid >> 5, lane = tid & 31;
  const int r = lane & 15, half = lane >> 4;     // lane owns row r, j-half
  bf16* aw = Attn + wave * 16 * ALD;             // this wave's attn tile
  const float st_r = stv[wave * 16 + r];
  const float bd   = b_dense[0];
  const int   j0   = half * 256;

  // Softmax pass 1: scores = tanh(...) in (-1,1) => no max-shift needed;
  // masked scores = -1e7 => exp underflows to 0 cleanly.
  float ssum = 0.f;
  for (int jj = 0; jj < 256; ++jj) {
    const int j = j0 + jj;
    const float sc = fast_tanhf(st_r + skv[j] + bd) + madd[j];
    ssum += __expf(sc);
  }
  ssum += __shfl_xor(ssum, 16, 32);              // merge the two j-halves
  const float rinv = 1.f / ssum;

  // Pass 2: write f32 attn to HBM, bf16 attn to LDS (WMMA A operand).
  float* ga = out_attn + ((size_t)u * I_ + ibase + wave * 16 + r) * J_;
  for (int jj = 0; jj < 256; ++jj) {
    const int j = j0 + jj;
    const float sc = fast_tanhf(st_r + skv[j] + bd) + madd[j];
    const float a = __expf(sc) * rinv;
    ga[j] = a;
    aw[r * ALD + j] = (bf16)a;
  }
  asm volatile("s_wait_dscnt 0" ::: "memory");   // attn tile visible wave-wide

  v8f vz = {0.f, 0.f, 0.f, 0.f, 0.f, 0.f, 0.f, 0.f};

  // ---- GEMM1: pooled(16x128) = attn(16x512) @ interacted(512x128), bf16 ----
  v8f acc[8];
#pragma unroll
  for (int n = 0; n < 8; ++n) acc[n] = vz;

  for (int k0 = 0; k0 < J_; k0 += 32) {
    // A frag (16-bit 16x32): lane<16 holds K 0..15, lane>=16 K 16..31
    Frag A;
    const bf16* ap = aw + r * ALD + k0 + half * 16;
    A.q[0] = *(const uint4*)ap;
    A.q[1] = *(const uint4*)(ap + 8);
#pragma unroll
    for (int n = 0; n < 8; ++n) {
      // B frag (16-bit 32x16): lane<16 -> K 0..15 col N=lane; lane>=16 -> K 16..31
      Frag B;
      const bf16* bp = Bt + (n * 16 + r) * JLD + k0 + half * 16;
      B.q[0] = *(const uint4*)bp;
      B.q[1] = *(const uint4*)(bp + 8);
      acc[n] = __builtin_amdgcn_wmma_f32_16x16x32_bf16(
          false, A.v, false, B.v, (short)0, acc[n], false, false);
    }
  }

  // ---- Reshape pooled C-layout -> A-layout via LDS (alias dead attn tile) --
  bf16* pool = aw;   // 16 x PLD bf16, fits inside the wave's attn region
#pragma unroll
  for (int n = 0; n < 8; ++n) {
#pragma unroll
    for (int p = 0; p < 8; ++p) {
      const int M = p + 8 * half;                // C/D layout: M = vgpr (+8 hi)
      pool[M * PLD + n * 16 + r] = (bf16)acc[n][p];
    }
  }
  asm volatile("s_wait_dscnt 0" ::: "memory");

  // ---- GEMM2: out(16x128) = pooled(16x128) @ W2(128x128), bf16 -------------
  v8f acc2[8];
#pragma unroll
  for (int n = 0; n < 8; ++n) acc2[n] = vz;

#pragma unroll
  for (int k0 = 0; k0 < C_; k0 += 32) {
    Frag A2;
    const bf16* ap2 = pool + r * PLD + k0 + half * 16;
    A2.q[0] = *(const uint4*)ap2;
    A2.q[1] = *(const uint4*)(ap2 + 8);
#pragma unroll
    for (int n = 0; n < 8; ++n) {
      Frag B2;
      const bf16* bp2 = W2t + (n * 16 + r) * W2LD + k0 + half * 16;
      B2.q[0] = *(const uint4*)bp2;
      B2.q[1] = *(const uint4*)(bp2 + 8);
      acc2[n] = __builtin_amdgcn_wmma_f32_16x16x32_bf16(
          false, A2.v, false, B2.v, (short)0, acc2[n], false, false);
    }
  }

  // ---- Epilogue: relu(upb + acc2), f32 store -------------------------------
  float* go = out_ctx + ((size_t)u * I_ + ibase + wave * 16) * C_;
#pragma unroll
  for (int n = 0; n < 8; ++n) {
    const float up = upb[n * 16 + r];
#pragma unroll
    for (int p = 0; p < 8; ++p) {
      const int M = p + 8 * half;
      const float v = acc2[n][p] + up;
      go[(size_t)M * C_ + n * 16 + r] = v > 0.f ? v : 0.f;
    }
  }
}

extern "C" void kernel_launch(void* const* d_in, const int* in_sizes, int n_in,
                              void* d_out, int out_size, void* d_ws, size_t ws_size,
                              hipStream_t stream) {
  const float*         target = (const float*)d_in[0];
  const float*         inter  = (const float*)d_in[1];
  const float*         user   = (const float*)d_in[2];
  const unsigned char* mask   = (const unsigned char*)d_in[3];  // bool array
  const float*         wdense = (const float*)d_in[4];
  const float*         bdense = (const float*)d_in[5];
  const float*         Wmlp   = (const float*)d_in[6];
  const float*         bmlp   = (const float*)d_in[7];

  float* out_ctx  = (float*)d_out;                          // (U,I,C) first
  float* out_attn = out_ctx + (size_t)U_ * I_ * C_;         // then (U,I,J)

  (void)in_sizes; (void)n_in; (void)out_size; (void)d_ws; (void)ws_size;

  // >48KB dynamic LDS opt-in (no-op under compile-only; capture-safe API).
  (void)hipFuncSetAttribute((const void*)ucap_kernel,
                            hipFuncAttributeMaxDynamicSharedMemorySize,
                            SMEM_BYTES);

  ucap_kernel<<<dim3(U_ * 2), dim3(256), SMEM_BYTES, stream>>>(
      target, inter, user, mask, wdense, bdense, Wmlp, bmlp, out_ctx, out_attn);
}
